// Filter_28278064677138
// MI455X (gfx1250) — matrix operand order repaired and checked by reading, not compiled, not run on hardware
//
#include <hip/hip_runtime.h>
#include <hip/hip_bf16.h>
#include <stdint.h>

// ---------------- problem constants ----------------
#define BATCH    4
#define D_MODEL  1024
#define SEQ_LEN  4096
#define EMB      5
#define ORDER    64
#define KREV_LEN (SEQ_LEN + 16)    // reversed k + 16 zero tail       = 4112
#define XPAD_LEN (SEQ_LEN + 256)   // 256 zero head + x               = 4352
#define KC_LEN   4120              // shifted-copy length (mult of 8)

typedef __bf16 v16bf __attribute__((ext_vector_type(16)));
typedef float  v8f   __attribute__((ext_vector_type(8)));

struct Q2 { uint4 a, b; };         // 32 B carrier for bit_cast -> v16bf

__device__ __forceinline__ unsigned short f32_to_bf16(float f) {
    unsigned u = __builtin_bit_cast(unsigned, f);
    u += 0x7FFFu + ((u >> 16) & 1u);     // round-to-nearest-even
    return (unsigned short)(u >> 16);
}
__device__ __forceinline__ __bf16 us_to_bf(unsigned short s) {
    return __builtin_bit_cast(__bf16, s);
}

// ============================================================
// Kernel 1: sin-MLP filter generator  h[l, 0..63]  (f32, tiny)
// ============================================================
__global__ void __launch_bounds__(256)
filter_mlp_kernel(const float* __restrict__ z,  const float* __restrict__ W1,
                  const float* __restrict__ b1, const float* __restrict__ freq,
                  const float* __restrict__ W2, const float* __restrict__ b2,
                  const float* __restrict__ W3, const float* __restrict__ b3,
                  float* __restrict__ h_out)
{
    __shared__ float sW1[ORDER * EMB], sb1[ORDER], sfr[ORDER];
    __shared__ float sW2[ORDER * ORDER], sb2[ORDER];
    __shared__ float sW3[ORDER * ORDER], sb3[ORDER];
    const int tid = threadIdx.x;
    for (int i = tid; i < ORDER * EMB; i += 256) sW1[i] = W1[i];
    for (int i = tid; i < ORDER; i += 256) {
        sb1[i] = b1[i]; sfr[i] = freq[i]; sb2[i] = b2[i]; sb3[i] = b3[i];
    }
    for (int i = tid; i < ORDER * ORDER; i += 256) { sW2[i] = W2[i]; sW3[i] = W3[i]; }
    __syncthreads();

    const int l = blockIdx.x * 256 + tid;
    if (l >= SEQ_LEN) return;

    float zv[EMB];
#pragma unroll
    for (int e = 0; e < EMB; ++e) zv[e] = z[l * EMB + e];

    float h1[ORDER], h2[ORDER];
#pragma unroll 4
    for (int o = 0; o < ORDER; ++o) {
        float a = sb1[o];
#pragma unroll
        for (int e = 0; e < EMB; ++e) a += zv[e] * sW1[o * EMB + e];
        h1[o] = __sinf(sfr[o] * a);
    }
#pragma unroll 4
    for (int p = 0; p < ORDER; ++p) {
        float a = sb2[p];
#pragma unroll 8
        for (int o = 0; o < ORDER; ++o) a += h1[o] * sW2[p * ORDER + o];
        h2[p] = __sinf(sfr[p] * a);
    }
#pragma unroll 4
    for (int p = 0; p < ORDER; ++p) {
        float a = sb3[p];
#pragma unroll 8
        for (int o = 0; o < ORDER; ++o) a += h2[o] * sW3[p * ORDER + o];
        h_out[l * ORDER + p] = __sinf(sfr[p] * a);
    }
}

// ============================================================
// Kernel 2: zero the 16-element tail pad of each reversed row
// ============================================================
__global__ void pad_zero_kernel(unsigned short* __restrict__ krev)
{
    int i = blockIdx.x * 256 + threadIdx.x;           // D_MODEL*16 entries
    if (i < D_MODEL * 16) {
        int d = i >> 4, r = i & 15;
        krev[d * KREV_LEN + SEQ_LEN + r] = 0;
    }
}

// ============================================================
// Kernel 3: k[d,l] = decay(d,l) * (W4 h^T)  via bf16 WMMA,
// written REVERSED in l. One 16x16 tile per wave; K=64 -> 2 WMMA.
// ============================================================
__global__ void __launch_bounds__(256)
filter_proj_kernel(const float* __restrict__ h,  const float* __restrict__ W4,
                   const float* __restrict__ t,  const float* __restrict__ deltas,
                   unsigned short* __restrict__ krev)
{
    const int wave = threadIdx.x >> 5;
    const int lane = threadIdx.x & 31;
    const int tile = blockIdx.x * 8 + wave;
    const int d0 = (tile >> 8) << 4;       // 64 d-tiles
    const int l0 = (tile & 255) << 4;      // 256 l-tiles
    const int half = lane >> 4;
    const int lq = lane & 15;

    // A = W4[d0+row, o]  (16x32 bf16, k = 16*half + e)
    v16bf a0, a1;
    {
        const int row = d0 + lq;
#pragma unroll
        for (int e = 0; e < 16; ++e) {
            const int koff = 16 * half + e;
            a0[e] = us_to_bf(f32_to_bf16(W4[row * ORDER + koff]));
            a1[e] = us_to_bf(f32_to_bf16(W4[row * ORDER + 32 + koff]));
        }
    }
    // B[o, l] = h[l, o]   (32x16 bf16, k = 16*half + e)
    v16bf bm0, bm1;
    {
        const int col = l0 + lq;
#pragma unroll
        for (int e = 0; e < 16; ++e) {
            const int k0 = 16 * half + e;
            bm0[e] = us_to_bf(f32_to_bf16(h[col * ORDER + k0]));
            bm1[e] = us_to_bf(f32_to_bf16(h[col * ORDER + 32 + k0]));
        }
    }
    v8f acc = {};
    acc = __builtin_amdgcn_wmma_f32_16x16x32_bf16(false, a0, false, bm0, (short)0, acc, false, false);
    acc = __builtin_amdgcn_wmma_f32_16x16x32_bf16(false, a1, false, bm1, (short)0, acc, false, false);

    // epilogue: decay + reversed write. VGPR r -> M(=d) = 8*half + r, N(=l) = lq.
    const int l = l0 + lq;
    const float tl = t[l];
    const int mbase = 8 * half;
#pragma unroll
    for (int r = 0; r < 8; ++r) {
        const int d = d0 + mbase + r;
        const float kv = acc[r] * __expf(-tl * fabsf(deltas[d]));
        krev[d * KREV_LEN + (SEQ_LEN - 1 - l)] = f32_to_bf16(kv);
    }
}

// ============================================================
// Kernel 4: causal long conv as implicit GEMM on WMMA.
// Block = one d channel; loops over b (krev LDS copies reused 4x).
//   Y[i,j] = y[l0 + 16 i + j]
//   A[i,n] = xpad[16 i + n + 16]        (aligned b128 x2 per lane)
//   B[n,j] = krev[cb2 - j + n],  cb2 = 3855 - l0
// B served from 8 element-shifted LDS copies (copy_r[i]=krev[i+r-8]).
// 8 waves; wave w owns tiles {w, 15-w} -> constant K work (4352).
// Two accumulator chains + unroll: 4 independent-ish WMMA per branch.
// Fused epilogue: out = y + x*bias[d].
// ============================================================
#define CONV_STEP(ACC, N0)                                                     \
    {                                                                          \
        const uint4* ap0 = (const uint4*)&s_x[abase + (N0)];                   \
        const uint4* ap1 = (const uint4*)&s_x[abase + (N0) + 16];              \
        Q2 qa; qa.a = ap0[0]; qa.b = ap1[0];                                   \
        const uint4* bp = (const uint4*)&kcopy[bbase + (N0)];                  \
        Q2 qb; qb.a = bp[0]; qb.b = bp[2];                                     \
        ACC = __builtin_amdgcn_wmma_f32_16x16x32_bf16(                         \
            false, __builtin_bit_cast(v16bf, qa), false,                       \
            __builtin_bit_cast(v16bf, qb), (short)0, ACC, false, false);       \
    }

__global__ void __launch_bounds__(256)
conv_kernel(const float* __restrict__ x, const unsigned short* __restrict__ krev,
            const float* __restrict__ bias, float* __restrict__ out)
{
    __shared__ unsigned short s_x[XPAD_LEN];        //  8,704 B
    __shared__ unsigned short s_kc[8 * KC_LEN];     // 65,920 B

    const int d = blockIdx.x;
    const int tid = threadIdx.x;
    const unsigned short* __restrict__ krow = krev + d * KREV_LEN;

    // krev copy 0 (global -> LDS), then shifted copies 1..7 (LDS -> LDS)
    for (int i = tid; i < KC_LEN; i += 256) {
        const int src = i - 8;
        s_kc[i] = ((unsigned)src < (unsigned)KREV_LEN) ? krow[src] : (unsigned short)0;
    }
    __syncthreads();
    for (int r = 1; r < 8; ++r)
        for (int i = tid; i < KC_LEN; i += 256)
            s_kc[r * KC_LEN + i] = (i + r < KC_LEN) ? s_kc[i + r] : (unsigned short)0;

    const int wave = tid >> 5, lane = tid & 31;
    const int half = lane >> 4, lq = lane & 15;
    const int rr = (7 - lq) & 7;                    // lane's residue -> copy select
    const unsigned short* __restrict__ kcopy = &s_kc[rr * KC_LEN];
    const float biasd = bias[d];

    for (int b = 0; b < BATCH; ++b) {
        const long row = ((long)b * D_MODEL + d) * SEQ_LEN;

        __syncthreads();   // prior K-loops done with s_x (and copies ready at b==0)
        for (int i = tid; i < XPAD_LEN; i += 256)
            s_x[i] = (i < 256) ? (unsigned short)0 : f32_to_bf16(x[row + (i - 256)]);
        __syncthreads();

        if (b + 1 < BATCH)  // warm L2/WGP$ for next batch row while we compute
            __builtin_prefetch(&x[row + (long)D_MODEL * SEQ_LEN], 0, 1);

        for (int pass = 0; pass < 2; ++pass) {
            const int tileIdx = pass ? (15 - wave) : wave;
            const int l0 = tileIdx << 8;
            const int Klen = l0 + 256;                          // multiple of 64
            const int cb2 = (SEQ_LEN - 1) - 240 - l0;           // 3855 - l0, ≡7 (mod 8)
            const int bbase = cb2 - lq + 8 * half + 8 - rr;     // ≡0 (mod 8)
            const int abase = 16 * lq + 16 + 8 * half;          // ≡0 (mod 8)

            v8f acc0 = {}, acc1 = {};
#pragma unroll 2
            for (int n0 = 0; n0 < Klen; n0 += 64) {
                CONV_STEP(acc0, n0);
                CONV_STEP(acc1, n0 + 32);
            }

            // epilogue: VGPR r -> M = 8*half + r, N = lq; pos = l0 + 16*M + N
            const long obase = row + l0 + 128 * half + lq;
#pragma unroll
            for (int r = 0; r < 8; ++r) {
                const long o = obase + 16 * r;
                out[o] = (acc0[r] + acc1[r]) + x[o] * biasd;
            }
        }
    }
}

// ============================================================
extern "C" void kernel_launch(void* const* d_in, const int* in_sizes, int n_in,
                              void* d_out, int out_size, void* d_ws, size_t ws_size,
                              hipStream_t stream)
{
    const float* x      = (const float*)d_in[0];
    // d_in[1] = L (scalar) — compile-time constant here
    const float* z      = (const float*)d_in[2];
    const float* t      = (const float*)d_in[3];
    const float* deltas = (const float*)d_in[4];
    const float* W1     = (const float*)d_in[5];
    const float* b1     = (const float*)d_in[6];
    const float* freq   = (const float*)d_in[7];
    const float* W2     = (const float*)d_in[8];
    const float* b2     = (const float*)d_in[9];
    const float* W3     = (const float*)d_in[10];
    const float* b3     = (const float*)d_in[11];
    const float* W4     = (const float*)d_in[12];
    const float* bias   = (const float*)d_in[13];
    float* out = (float*)d_out;

    // workspace layout: h (f32, 1 MB) | krev (bf16, 8.4 MB)
    char* ws = (char*)d_ws;
    float* h = (float*)ws;
    size_t h_bytes = (((size_t)SEQ_LEN * ORDER * sizeof(float)) + 255) & ~(size_t)255;
    unsigned short* krev = (unsigned short*)(ws + h_bytes);

    filter_mlp_kernel<<<SEQ_LEN / 256, 256, 0, stream>>>(z, W1, b1, freq, W2, b2, W3, b3, h);
    pad_zero_kernel<<<(D_MODEL * 16 + 255) / 256, 256, 0, stream>>>(krev);
    filter_proj_kernel<<<(D_MODEL / 16) * (SEQ_LEN / 16) / 8, 256, 0, stream>>>(h, W4, t, deltas, krev);
    conv_kernel<<<D_MODEL, 256, 0, stream>>>(x, krev, bias, out);
}
